// RNNTracker_49821620634041
// MI455X (gfx1250) — matrix operand order repaired and checked by reading, compiled, not run on hardware
//
#include <hip/hip_runtime.h>
#include <hip/hip_bf16.h>
#include <stdint.h>

// ---------------------------------------------------------------------------
// Problem constants (reference: B=1024, T=256, D=128, H=512, F=256)
// ---------------------------------------------------------------------------
#define BB   1024
#define TT   256
#define DD   128
#define HH   512
#define FF   256
#define G4H  (4*HH)          // 2048 gate columns
#define KTOT (DD+HH)         // 640 fused K for the step GEMM
#define KC_STEP (KTOT/32)    // 20 k-chunks of 32
#define KC_FEAT (HH/32)      // 16 k-chunks

typedef __attribute__((ext_vector_type(16))) __bf16 v16bf;
typedef __attribute__((ext_vector_type(8)))  float  v8f;

union Frag16 {               // one WMMA bf16 operand (16 bf16 per lane = 32B)
    v16bf v;
    unsigned short u[16];
    uint4 q[2];
};

__device__ __forceinline__ unsigned short f2bf(float f) {
    unsigned u = __float_as_uint(f);
    u += 0x7FFFu + ((u >> 16) & 1u);   // round-to-nearest-even
    return (unsigned short)(u >> 16);
}
__device__ __forceinline__ float sigmoid_fast(float x) {
    // 1/(1+exp(-x)) with v_exp_f32 + v_rcp_f32
    return __builtin_amdgcn_rcpf(1.0f + __expf(-x));
}
__device__ __forceinline__ float tanh_fast(float x) {
    // tanh(x) = sign(x) * (1-e)/(1+e), e = exp(-2|x|); safe at large |x|
    float e = __expf(-2.0f * __builtin_fabsf(x));
    float r = (1.0f - e) * __builtin_amdgcn_rcpf(1.0f + e);
    return __builtin_copysignf(r, x);
}

// CDNA5 direct global->LDS async copy (ASYNCcnt), GVS addressing:
//   LDS address from the vdst VGPR, global address = SGPR base + voffset
__device__ __forceinline__ void async_b128(unsigned lds_addr, unsigned voff,
                                           const void* base) {
    asm volatile("global_load_async_to_lds_b128 %0, %1, %2"
                 :: "v"(lds_addr), "v"(voff), "s"(base)
                 : "memory");
}
__device__ __forceinline__ void wait_async0() {
    asm volatile("s_wait_asynccnt 0x0" ::: "memory");
}
__device__ __forceinline__ unsigned lds_off(const void* p) {
    // generic pointer to LDS: flat addr low 32 bits == LDS byte address
    return (unsigned)(uintptr_t)p;
}

// ---------------------------------------------------------------------------
// Prep: pack [W_ih | W_hh] (2048 x 640, fp32) into bf16 WMMA B-fragment tiles.
// Tile (kc, nt) = 32x16 block of B[k][g] = W[g][k]; per lane l: 16 K values of
// column n=l%16, K = 8*(l/16)+{0..7, 16..23} (table order).  1 KB per tile.
// ---------------------------------------------------------------------------
__global__ void pack_gates_kernel(const float* __restrict__ W_ih,
                                  const float* __restrict__ W_hh,
                                  unsigned short* __restrict__ Wp) {
    int idx = blockIdx.x * 256 + threadIdx.x;
    if (idx >= KC_STEP * (G4H/16) * 512) return;
    int e    = idx & 15;          // element within lane (K direction)
    int l    = (idx >> 4) & 31;   // lane
    int tile = idx >> 9;
    int kc   = tile >> 7;         // tile / 128
    int nt   = tile & 127;
    int g = nt * 16 + (l & 15);
    int k = kc * 32 + ((l >= 16) ? 8 : 0) + (e & 7) + ((e >= 8) ? 16 : 0);
    float val = (k < DD) ? W_ih[(size_t)g * DD + k]
                         : W_hh[(size_t)g * HH + (k - DD)];
    Wp[idx] = f2bf(val);
}

__global__ void pack_feat_kernel(const float* __restrict__ W_feat,
                                 unsigned short* __restrict__ Wfp) {
    int idx = blockIdx.x * 256 + threadIdx.x;
    if (idx >= KC_FEAT * (FF/16) * 512) return;
    int e    = idx & 15;
    int l    = (idx >> 4) & 31;
    int tile = idx >> 9;
    int kc   = tile >> 4;         // tile / 16
    int nt   = tile & 15;
    int g = nt * 16 + (l & 15);
    int k = kc * 32 + ((l >= 16) ? 8 : 0) + (e & 7) + ((e >= 8) ? 16 : 0);
    Wfp[idx] = f2bf(W_feat[(size_t)g * HH + k]);
}

// Zero h double-buffers + c, combine biases.
__global__ void init_kernel(const float* __restrict__ b_ih,
                            const float* __restrict__ b_hh,
                            float* __restrict__ bias,
                            unsigned short* __restrict__ h0,
                            unsigned short* __restrict__ h1,
                            float* __restrict__ c) {
    int i = blockIdx.x * 256 + threadIdx.x;
    if (i < BB * HH) { h0[i] = 0; h1[i] = 0; c[i] = 0.0f; }
    if (i < G4H)     { bias[i] = b_ih[i] + b_hh[i]; }
}

// ---------------------------------------------------------------------------
// One LSTM timestep, fused GEMM + gates.
// Grid (8, 16): blockIdx.x = 128-row slab, blockIdx.y = 32-wide j-column slab.
// 256 threads = 8 waves; wave w owns rows r0..r0+15 and all 4 gate tiles for
// its 32 j-columns -> 8 accumulators (2 j-tiles x 4 gates).
// ---------------------------------------------------------------------------
__global__ __launch_bounds__(256)
void lstm_step_kernel(const float* __restrict__ x,          // [B,T,D]
                      const unsigned short* __restrict__ Wp,// packed gates B
                      const float* __restrict__ bias,       // [4H]
                      const unsigned short* __restrict__ h_in,  // [B,H] bf16
                      unsigned short* __restrict__ h_out,       // [B,H] bf16
                      float* __restrict__ c,                // [B,H] fp32
                      int t) {
    __shared__ unsigned short ldsB[2][8 * 512];  // 2 x 8 tiles x 1 KB

    const int tid  = threadIdx.x;
    const int lane = tid & 31;
    const int w    = tid >> 5;
    const int jb   = blockIdx.y;                 // 0..15
    const int j0   = jb * 32;
    const int r0   = blockIdx.x * 128 + w * 16;
    const int mrow = r0 + (lane & 15);
    const int koff = (lane < 16) ? 0 : 8;

    // direct async copy of 8 B-tiles (4 gates x 2 j-tiles) for chunk kc
    auto copyB = [&](int kc, int buf) {
        int gi = tid >> 6;                       // 0..3
        int q  = tid & 63;                       // 0..63
        unsigned src = (unsigned)(((kc * 128 + gi * 32 + jb * 2) * 512 + q * 16) * 2);
        unsigned dst = lds_off(&ldsB[buf][gi * 1024 + q * 16]);
        async_b128(dst, src, Wp);
        async_b128(dst + 16, src + 16, Wp);
    };
    auto loadBfrag = [&](Frag16& b, int cur, int ti) {
        const unsigned short* bp = &ldsB[cur][ti * 512 + lane * 16];
        b.q[0] = ((const uint4*)bp)[0];
        b.q[1] = ((const uint4*)bp)[1];
    };

    v8f acc[8];
#pragma unroll
    for (int i = 0; i < 8; ++i)
#pragma unroll
        for (int v = 0; v < 8; ++v) acc[i][v] = 0.0f;

    copyB(0, 0);
    wait_async0();
    __syncthreads();

    for (int kc = 0; kc < KC_STEP; ++kc) {
        const int cur = kc & 1;
        if (kc + 1 < KC_STEP) copyB(kc + 1, cur ^ 1);   // overlaps WMMA chain

        // ---- A fragment: rows = mrow, K = kc*32 + {koff..koff+7, koff+16..}
        Frag16 a;
        const int k0 = kc * 32;
        if (kc < DD / 32) {                      // x region (fp32 -> bf16)
            const float* xr = x + ((size_t)mrow * TT + t) * DD + k0 + koff;
            float4 f0 = *(const float4*)(xr + 0);
            float4 f1 = *(const float4*)(xr + 4);
            float4 f2 = *(const float4*)(xr + 16);
            float4 f3 = *(const float4*)(xr + 20);
            float xf[16] = {f0.x, f0.y, f0.z, f0.w, f1.x, f1.y, f1.z, f1.w,
                            f2.x, f2.y, f2.z, f2.w, f3.x, f3.y, f3.z, f3.w};
#pragma unroll
            for (int i = 0; i < 16; ++i) a.u[i] = f2bf(xf[i]);
        } else {                                 // h region (already bf16)
            const unsigned short* hr =
                h_in + (size_t)mrow * HH + (k0 - DD) + koff;
            a.q[0] = *(const uint4*)(hr);
            a.q[1] = *(const uint4*)(hr + 16);
        }

        // ---- 8 tiles: pipelined B fragments from LDS, WMMA accumulate
        Frag16 bf[2];
        loadBfrag(bf[0], cur, 0);
#pragma unroll
        for (int ti = 0; ti < 8; ++ti) {
            if (ti < 7) loadBfrag(bf[(ti + 1) & 1], cur, ti + 1);
            acc[ti] = __builtin_amdgcn_wmma_f32_16x16x32_bf16(
                false, a.v, false, bf[ti & 1].v, (short)0, acc[ti], false, false);
        }
        wait_async0();         // next-buffer async copy landed in LDS
        __syncthreads();
    }

    // ---- fused LSTM pointwise epilogue (all four gates are in registers)
    const int n      = lane & 15;
    const int rowAdd = (lane >> 4) * 8;
#pragma unroll
    for (int tj = 0; tj < 2; ++tj) {
        const int j = j0 + tj * 16 + n;          // 0..511
        const float bi = bias[j];
        const float bf_ = bias[HH + j];
        const float bg = bias[2 * HH + j];
        const float bo = bias[3 * HH + j];
#pragma unroll
        for (int v = 0; v < 8; ++v) {
            const int row = r0 + v + rowAdd;
            const float ig = sigmoid_fast(acc[0 + tj][v] + bi);
            const float fg = sigmoid_fast(acc[2 + tj][v] + bf_);
            const float gg = tanh_fast(acc[4 + tj][v] + bg);
            const float og = sigmoid_fast(acc[6 + tj][v] + bo);
            const size_t idx = (size_t)row * HH + j;
            const float cn = fg * c[idx] + ig * gg;
            c[idx] = cn;
            h_out[idx] = f2bf(og * tanh_fast(cn));
        }
    }
}

// ---------------------------------------------------------------------------
// features = h_T @ W_feat^T + b_feat   (1024 x 256, K = 512), WMMA bf16.
// Grid (8, 4): 128-row slab x 64-col slab; 8 waves x (16 rows, 4 tiles).
// ---------------------------------------------------------------------------
__global__ __launch_bounds__(256)
void feat_kernel(const unsigned short* __restrict__ h,     // [B,H] bf16
                 const unsigned short* __restrict__ Wfp,   // packed feat B
                 const float* __restrict__ b_feat,
                 float* __restrict__ feat) {               // [B,F] fp32
    __shared__ unsigned short ldsB[2][4 * 512];            // 2 x 4 KB

    const int tid  = threadIdx.x;
    const int lane = tid & 31;
    const int w    = tid >> 5;
    const int jb   = blockIdx.y;                 // 0..3
    const int j0   = jb * 64;
    const int r0   = blockIdx.x * 128 + w * 16;
    const int mrow = r0 + (lane & 15);
    const int koff = (lane < 16) ? 0 : 8;

    auto copyB = [&](int kc, int buf) {
        unsigned src = (unsigned)(((kc * 16 + jb * 4) * 512 + tid * 8) * 2);
        unsigned dst = lds_off(&ldsB[buf][tid * 8]);
        async_b128(dst, src, Wfp);
    };
    auto loadBfrag = [&](Frag16& b, int cur, int ti) {
        const unsigned short* bp = &ldsB[cur][ti * 512 + lane * 16];
        b.q[0] = ((const uint4*)bp)[0];
        b.q[1] = ((const uint4*)bp)[1];
    };

    v8f acc[4];
#pragma unroll
    for (int i = 0; i < 4; ++i)
#pragma unroll
        for (int v = 0; v < 8; ++v) acc[i][v] = 0.0f;

    copyB(0, 0);
    wait_async0();
    __syncthreads();

    for (int kc = 0; kc < KC_FEAT; ++kc) {
        const int cur = kc & 1;
        if (kc + 1 < KC_FEAT) copyB(kc + 1, cur ^ 1);

        Frag16 a;
        const unsigned short* hr = h + (size_t)mrow * HH + kc * 32 + koff;
        a.q[0] = *(const uint4*)(hr);
        a.q[1] = *(const uint4*)(hr + 16);

        Frag16 bf[2];
        loadBfrag(bf[0], cur, 0);
#pragma unroll
        for (int ti = 0; ti < 4; ++ti) {
            if (ti < 3) loadBfrag(bf[(ti + 1) & 1], cur, ti + 1);
            acc[ti] = __builtin_amdgcn_wmma_f32_16x16x32_bf16(
                false, a.v, false, bf[ti & 1].v, (short)0, acc[ti], false, false);
        }
        wait_async0();
        __syncthreads();
    }

    const int n      = lane & 15;
    const int rowAdd = (lane >> 4) * 8;
#pragma unroll
    for (int ti = 0; ti < 4; ++ti) {
        const int col = j0 + ti * 16 + n;
        const float bf_ = b_feat[col];
#pragma unroll
        for (int v = 0; v < 8; ++v) {
            const int row = r0 + v + rowAdd;
            feat[(size_t)row * FF + col] = acc[ti][v] + bf_;
        }
    }
}

// out = features @ W_out^T + b_out   (N=2: plain dot per row)
__global__ void head_kernel(const float* __restrict__ feat,
                            const float* __restrict__ W_out,  // [2,F]
                            const float* __restrict__ b_out,
                            float* __restrict__ out) {        // [B,2]
    int row = blockIdx.x * blockDim.x + threadIdx.x;
    if (row >= BB) return;
    float a0 = b_out[0], a1 = b_out[1];
    const float* fr = feat + (size_t)row * FF;
    for (int f = 0; f < FF; ++f) {
        float v = fr[f];
        a0 += v * W_out[f];
        a1 += v * W_out[FF + f];
    }
    out[row * 2 + 0] = a0;
    out[row * 2 + 1] = a1;
}

// ---------------------------------------------------------------------------
extern "C" void kernel_launch(void* const* d_in, const int* in_sizes, int n_in,
                              void* d_out, int out_size, void* d_ws, size_t ws_size,
                              hipStream_t stream) {
    const float* x      = (const float*)d_in[0];   // isequence [B,T,D]
    const float* W_ih   = (const float*)d_in[1];
    const float* W_hh   = (const float*)d_in[2];
    const float* b_ih   = (const float*)d_in[3];
    const float* b_hh   = (const float*)d_in[4];
    const float* W_feat = (const float*)d_in[5];
    const float* b_feat = (const float*)d_in[6];
    const float* W_out  = (const float*)d_in[7];
    const float* b_out  = (const float*)d_in[8];
    float* out = (float*)d_out;

    // workspace carve-up (all sizes multiple of 256B)
    uint8_t* p = (uint8_t*)d_ws;
    unsigned short* Wp   = (unsigned short*)p; p += (size_t)KC_STEP * 128 * 512 * 2; // 2.62 MB
    unsigned short* Wfp  = (unsigned short*)p; p += (size_t)KC_FEAT * 16  * 512 * 2; // 256 KB
    float*          bias = (float*)p;          p += (size_t)G4H * 4;                  // 8 KB
    unsigned short* h0   = (unsigned short*)p; p += (size_t)BB * HH * 2;              // 1 MB
    unsigned short* h1   = (unsigned short*)p; p += (size_t)BB * HH * 2;              // 1 MB
    float*          c    = (float*)p;          p += (size_t)BB * HH * 4;              // 2 MB
    float*          feat = (float*)p;          p += (size_t)BB * FF * 4;              // 1 MB

    // prep (re-run every call: deterministic, graph-capture safe)
    pack_gates_kernel<<<(KC_STEP * 128 * 512 + 255) / 256, 256, 0, stream>>>(W_ih, W_hh, Wp);
    pack_feat_kernel<<<(KC_FEAT * 16 * 512 + 255) / 256, 256, 0, stream>>>(W_feat, Wfp);
    init_kernel<<<(BB * HH + 255) / 256, 256, 0, stream>>>(b_ih, b_hh, bias, h0, h1, c);

    // recurrence: h ping-pongs; t even reads h0 writes h1, t odd reads h1 writes h0
    dim3 gstep(8, 16);
    for (int t = 0; t < TT; ++t) {
        const unsigned short* hin = (t & 1) ? h1 : h0;
        unsigned short*      hout = (t & 1) ? h0 : h1;
        lstm_step_kernel<<<gstep, 256, 0, stream>>>(x, Wp, bias, hin, hout, c, t);
    }
    // T=256 even -> final h landed in h0

    feat_kernel<<<dim3(8, 4), 256, 0, stream>>>(h0, Wfp, b_feat, feat);
    head_kernel<<<(BB + 255) / 256, 256, 0, stream>>>(feat, W_out, b_out, out);
}
